// MyRNN_54460185313601
// MI455X (gfx1250) — matrix operand order repaired and checked by reading, not compiled, not run on hardware
//
#include <hip/hip_runtime.h>
#include <math.h>

typedef __attribute__((ext_vector_type(16))) __bf16          v16bf;
typedef __attribute__((ext_vector_type(16))) unsigned short  v16us;
typedef __attribute__((ext_vector_type(8)))  float           v8f;

#define SEQ    512
#define EMB    128
#define UNITS  128
#define ROWS   16                 // batch rows per workgroup
#define BATCH  256
#define NBLK   (BATCH / ROWS)     // 16 recurrence workgroups
#define TSTEPS 16                 // timesteps per phase-1 workgroup
#define TILE   (ROWS * UNITS)     // 2048 elements per (t, blk) tile

#if __has_builtin(__builtin_amdgcn_tanhf)
#define TANH(x) __builtin_amdgcn_tanhf(x)
#else
#define TANH(x) tanhf(x)
#endif

// ---- conversions -----------------------------------------------------------
// RNE (cold path: one-time weight conversion)
__device__ __forceinline__ unsigned short f2bf_rne(float f) {
    unsigned int u = __builtin_bit_cast(unsigned int, f);
    u += 0x7FFFu + ((u >> 16) & 1u);
    return (unsigned short)(u >> 16);
}
// truncation (hot path: 1 VALU op, error irrelevant post-tanh)
__device__ __forceinline__ unsigned short f2bf_tr(float f) {
    return (unsigned short)(__builtin_bit_cast(unsigned int, f) >> 16);
}

// ---- fragment helpers ------------------------------------------------------
// A-fragment LDS layout (16x32 bf16 tile per K-block kb):
//   chunk(kb, half): 16 rows x 16 contiguous bf16 (32B) per lane.
//   half 0 (lanes 0..15)  holds K%32 in {0..7 -> j0..7, 16..23 -> j8..15}
//   half 1 (lanes 16..31) holds K%32 in {8..15 -> j0..7, 24..31 -> j8..15}
__device__ __forceinline__ v16bf load_afrag(const unsigned short* buf, int kb, int lane) {
    const int half = (lane < 16) ? 0 : 1;
    const int row  = lane & 15;
    const int idx  = ((kb * 2 + half) * 16 + row) * 16;
    return *reinterpret_cast<const v16bf*>(buf + idx);
}

// tanh(acc) -> bf16 (trunc), scattered into A-fragment layout.
// C/D layout: VGPR r holds M = r + (lane<16?0:8), N = nbase + (lane&15).
__device__ __forceinline__ void store_htile(unsigned short* buf, v8f acc, int nbase, int lane) {
    const int n    = nbase + (lane & 15);
    const int kb   = n >> 5;
    const int rk   = n & 31;
    const int half = (rk >> 3) & 1;
    const int j    = (rk & 7) + ((rk >= 16) ? 8 : 0);
    const int mofs = (lane < 16) ? 0 : 8;
    const int base = (kb * 2 + half) * 256 + j;
#pragma unroll
    for (int r = 0; r < 8; ++r)
        buf[base + (r + mofs) * 16] = f2bf_tr(TANH(acc[r]));
}

// B-fragment (32x16) of a 128x128 row-major f32 matrix -> registers (bf16).
__device__ __forceinline__ v16bf load_bfrag(const float* __restrict__ W, int kb, int nbase, int lane) {
    const int n    = nbase + (lane & 15);
    const int kofs = kb * 32 + ((lane < 16) ? 0 : 8);
    v16us w;
#pragma unroll
    for (int j = 0; j < 16; ++j) w[j] = f2bf_rne(W[(kofs + j + ((j >= 8) ? 8 : 0)) * UNITS + n]);
    return __builtin_bit_cast(v16bf, w);
}

// ============================================================================
// Phase 1: XW[t, blk, n, m] = (emb[inputs] @ w0x + b0)   — fully parallel GEMM
// Tile layout: [n 0..127][M 0..15] so each lane reads/writes 8 consecutive f32.
// ============================================================================
__global__ __launch_bounds__(256)
void xw_precompute_kernel(const int* __restrict__ inputs, const float* __restrict__ emb,
                          const float* __restrict__ w0x, const float* __restrict__ b0v,
                          float* __restrict__ xw)
{
    __shared__ __align__(32) unsigned short x_lds[TILE];

    const int tid   = threadIdx.x;
    const int lane  = tid & 31;
    const int wave  = tid >> 5;
    const int nbase = wave * 16;
    const int blk   = blockIdx.x;
    const int brow0 = blk * ROWS;
    const int t0    = blockIdx.y * TSTEPS;

    v16bf Wx[4];
#pragma unroll
    for (int kb = 0; kb < 4; ++kb) Wx[kb] = load_bfrag(w0x, kb, nbase, lane);
    const float bias0 = b0v[nbase + (lane & 15)];

    const int xrow = tid & 15;
    const int xa   = tid >> 4;
    const int xkb = xa >> 2, xhalf = xa & 1, xjoff = (xa & 2) ? 8 : 0;
    const int xdst = ((xkb * 2 + xhalf) * 16 + xrow) * 16 + xjoff;
    const int* tokrow = inputs + (size_t)(brow0 + xrow) * SEQ;

    const int n    = nbase + (lane & 15);
    const int mofs = (lane < 16) ? 0 : 8;

    for (int tt = 0; tt < TSTEPS; ++tt) {
        const int t = t0 + tt;
        {   // gather x_t -> bf16 A-layout
            const float4* src = reinterpret_cast<const float4*>(emb + (size_t)tokrow[t] * EMB + xa * 8);
            float4 f0 = src[0], f1 = src[1];
            if (tt + 1 < TSTEPS)
                __builtin_prefetch(emb + (size_t)tokrow[t + 1] * EMB + xa * 8, 0, 1);
            union { unsigned short u16[8]; uint4 v; } pk;
            pk.u16[0] = f2bf_tr(f0.x); pk.u16[1] = f2bf_tr(f0.y);
            pk.u16[2] = f2bf_tr(f0.z); pk.u16[3] = f2bf_tr(f0.w);
            pk.u16[4] = f2bf_tr(f1.x); pk.u16[5] = f2bf_tr(f1.y);
            pk.u16[6] = f2bf_tr(f1.z); pk.u16[7] = f2bf_tr(f1.w);
            *reinterpret_cast<uint4*>(&x_lds[xdst]) = pk.v;
        }
        __syncthreads();

        v8f acc = { bias0, bias0, bias0, bias0, bias0, bias0, bias0, bias0 };
#pragma unroll
        for (int kb = 0; kb < 4; ++kb) {
            v16bf a = load_afrag(x_lds, kb, lane);
            acc = __builtin_amdgcn_wmma_f32_16x16x32_bf16(false, a, false, Wx[kb],
                                                          (short)0, acc, false, false);
        }
        __syncthreads();   // x_lds reads done before next gather overwrites

        float* dst = xw + ((size_t)t * NBLK + blk) * TILE + n * ROWS + mofs;
        float4 lo = { acc[0], acc[1], acc[2], acc[3] };
        float4 hi = { acc[4], acc[5], acc[6], acc[7] };
        *reinterpret_cast<float4*>(dst)     = lo;
        *reinterpret_cast<float4*>(dst + 4) = hi;
    }
}

// ============================================================================
// Phase 2: sequential recurrence; acc0 starts from precomputed XW tile.
// ============================================================================
__global__ __launch_bounds__(256)
void rnn2_recur_kernel(const float* __restrict__ xw,
                       const float* __restrict__ w0h,
                       const float* __restrict__ w1x, const float* __restrict__ w1h,
                       const float* __restrict__ b1v,
                       const float* __restrict__ wd1, const float* __restrict__ bd1,
                       const float* __restrict__ wd2, const float* __restrict__ bd2,
                       float* __restrict__ out)
{
    __shared__ __align__(32) unsigned short h0_lds[2][TILE];
    __shared__ __align__(32) unsigned short h1_lds[2][TILE];
    __shared__ float z_lds[ROWS][UNITS];

    const int tid   = threadIdx.x;
    const int lane  = tid & 31;
    const int wave  = tid >> 5;
    const int nbase = wave * 16;
    const int blk   = blockIdx.x;
    const int brow0 = blk * ROWS;

    v16bf W0h[4], W1x[4], W1h[4];
#pragma unroll
    for (int kb = 0; kb < 4; ++kb) {
        W0h[kb] = load_bfrag(w0h, kb, nbase, lane);
        W1x[kb] = load_bfrag(w1x, kb, nbase, lane);
        W1h[kb] = load_bfrag(w1h, kb, nbase, lane);
    }
    const float bias1 = b1v[nbase + (lane & 15)];

    for (int i = tid; i < TILE; i += 256) { h0_lds[0][i] = 0; h1_lds[0][i] = 0; }

    const int n    = nbase + (lane & 15);
    const int mofs = (lane < 16) ? 0 : 8;
    const float* xsrc = xw + (size_t)blk * TILE + n * ROWS + mofs;

    __syncthreads();

    for (int t = 0; t < SEQ; ++t) {
        const int p = t & 1;

        // ---- layer 0: h0_new = tanh(XW[t] + h0_old @ w0h) ----
        const float* xs = xsrc + (size_t)t * (NBLK * TILE);
        float4 lo = *reinterpret_cast<const float4*>(xs);
        float4 hi = *reinterpret_cast<const float4*>(xs + 4);
        if (t + 1 < SEQ)
            __builtin_prefetch(xs + (size_t)(NBLK * TILE), 0, 1);
        v8f acc0 = { lo.x, lo.y, lo.z, lo.w, hi.x, hi.y, hi.z, hi.w };
#pragma unroll
        for (int kb = 0; kb < 4; ++kb) {
            v16bf a = load_afrag(h0_lds[p], kb, lane);
            acc0 = __builtin_amdgcn_wmma_f32_16x16x32_bf16(false, a, false, W0h[kb],
                                                           (short)0, acc0, false, false);
        }
        store_htile(h0_lds[p ^ 1], acc0, nbase, lane);
        __syncthreads();

        // ---- layer 1: two independent WMMA chains, summed ----
        v8f accA = { 0.f, 0.f, 0.f, 0.f, 0.f, 0.f, 0.f, 0.f };
        v8f accB = { bias1, bias1, bias1, bias1, bias1, bias1, bias1, bias1 };
#pragma unroll
        for (int kb = 0; kb < 4; ++kb) {
            v16bf aA = load_afrag(h0_lds[p ^ 1], kb, lane);
            accA = __builtin_amdgcn_wmma_f32_16x16x32_bf16(false, aA, false, W1x[kb],
                                                           (short)0, accA, false, false);
            v16bf aB = load_afrag(h1_lds[p], kb, lane);
            accB = __builtin_amdgcn_wmma_f32_16x16x32_bf16(false, aB, false, W1h[kb],
                                                           (short)0, accB, false, false);
        }
        store_htile(h1_lds[p ^ 1], accA + accB, nbase, lane);
        __syncthreads();
    }

    // ---- MLP head ----
    const unsigned short* h1f = h1_lds[SEQ & 1];
    {
        const float biasd = bd1[n];
        v8f accz = { biasd, biasd, biasd, biasd, biasd, biasd, biasd, biasd };
#pragma unroll
        for (int kb = 0; kb < 4; ++kb) {
            v16bf a  = load_afrag(h1f, kb, lane);
            v16bf wb = load_bfrag(wd1, kb, nbase, lane);
            accz = __builtin_amdgcn_wmma_f32_16x16x32_bf16(false, a, false, wb,
                                                           (short)0, accz, false, false);
        }
#pragma unroll
        for (int r = 0; r < 8; ++r)
            z_lds[r + mofs][n] = fmaxf(accz[r], 0.0f);
    }
    __syncthreads();

    if (tid < ROWS) {
        float s = bd2[0];
#pragma unroll 4
        for (int nn = 0; nn < UNITS; ++nn) s += z_lds[tid][nn] * wd2[nn];
        out[brow0 + tid] = 1.0f / (1.0f + expf(-s));
    }
}

// ============================================================================
// Fallback: fully fused single kernel (used when workspace is too small).
// ============================================================================
__global__ __launch_bounds__(256)
void rnn2_fused_kernel(const int* __restrict__ inputs, const float* __restrict__ emb,
                       const float* __restrict__ w0x, const float* __restrict__ w0h,
                       const float* __restrict__ b0v,
                       const float* __restrict__ w1x, const float* __restrict__ w1h,
                       const float* __restrict__ b1v,
                       const float* __restrict__ wd1, const float* __restrict__ bd1,
                       const float* __restrict__ wd2, const float* __restrict__ bd2,
                       float* __restrict__ out)
{
    __shared__ __align__(32) unsigned short x_lds[TILE];
    __shared__ __align__(32) unsigned short h0_lds[2][TILE];
    __shared__ __align__(32) unsigned short h1_lds[2][TILE];
    __shared__ float z_lds[ROWS][UNITS];

    const int tid   = threadIdx.x;
    const int lane  = tid & 31;
    const int wave  = tid >> 5;
    const int nbase = wave * 16;
    const int brow0 = blockIdx.x * ROWS;

    v16bf W0x[4], W0h[4], W1x[4], W1h[4];
#pragma unroll
    for (int kb = 0; kb < 4; ++kb) {
        W0x[kb] = load_bfrag(w0x, kb, nbase, lane);
        W0h[kb] = load_bfrag(w0h, kb, nbase, lane);
        W1x[kb] = load_bfrag(w1x, kb, nbase, lane);
        W1h[kb] = load_bfrag(w1h, kb, nbase, lane);
    }
    const float bias0 = b0v[nbase + (lane & 15)];
    const float bias1 = b1v[nbase + (lane & 15)];

    for (int i = tid; i < TILE; i += 256) { h0_lds[0][i] = 0; h1_lds[0][i] = 0; }

    const int xrow = tid & 15;
    const int xa   = tid >> 4;
    const int xkb = xa >> 2, xhalf = xa & 1, xjoff = (xa & 2) ? 8 : 0;
    const int xdst = ((xkb * 2 + xhalf) * 16 + xrow) * 16 + xjoff;
    const int* tokrow = inputs + (size_t)(brow0 + xrow) * SEQ;

    __syncthreads();

    for (int t = 0; t < SEQ; ++t) {
        const int p = t & 1;
        {
            const float4* src = reinterpret_cast<const float4*>(emb + (size_t)tokrow[t] * EMB + xa * 8);
            float4 f0 = src[0], f1 = src[1];
            if (t + 1 < SEQ)
                __builtin_prefetch(emb + (size_t)tokrow[t + 1] * EMB + xa * 8, 0, 1);
            union { unsigned short u16[8]; uint4 v; } pk;
            pk.u16[0] = f2bf_tr(f0.x); pk.u16[1] = f2bf_tr(f0.y);
            pk.u16[2] = f2bf_tr(f0.z); pk.u16[3] = f2bf_tr(f0.w);
            pk.u16[4] = f2bf_tr(f1.x); pk.u16[5] = f2bf_tr(f1.y);
            pk.u16[6] = f2bf_tr(f1.z); pk.u16[7] = f2bf_tr(f1.w);
            *reinterpret_cast<uint4*>(&x_lds[xdst]) = pk.v;
        }
        __syncthreads();

        v8f accX = { bias0, bias0, bias0, bias0, bias0, bias0, bias0, bias0 };
        v8f accH = { 0.f, 0.f, 0.f, 0.f, 0.f, 0.f, 0.f, 0.f };
#pragma unroll
        for (int kb = 0; kb < 4; ++kb) {
            v16bf ax = load_afrag(x_lds, kb, lane);
            accX = __builtin_amdgcn_wmma_f32_16x16x32_bf16(false, ax, false, W0x[kb],
                                                           (short)0, accX, false, false);
            v16bf ah = load_afrag(h0_lds[p], kb, lane);
            accH = __builtin_amdgcn_wmma_f32_16x16x32_bf16(false, ah, false, W0h[kb],
                                                           (short)0, accH, false, false);
        }
        store_htile(h0_lds[p ^ 1], accX + accH, nbase, lane);
        __syncthreads();

        v8f accA = { 0.f, 0.f, 0.f, 0.f, 0.f, 0.f, 0.f, 0.f };
        v8f accB = { bias1, bias1, bias1, bias1, bias1, bias1, bias1, bias1 };
#pragma unroll
        for (int kb = 0; kb < 4; ++kb) {
            v16bf aA = load_afrag(h0_lds[p ^ 1], kb, lane);
            accA = __builtin_amdgcn_wmma_f32_16x16x32_bf16(false, aA, false, W1x[kb],
                                                           (short)0, accA, false, false);
            v16bf aB = load_afrag(h1_lds[p], kb, lane);
            accB = __builtin_amdgcn_wmma_f32_16x16x32_bf16(false, aB, false, W1h[kb],
                                                           (short)0, accB, false, false);
        }
        store_htile(h1_lds[p ^ 1], accA + accB, nbase, lane);
        __syncthreads();
    }

    const unsigned short* h1f = h1_lds[SEQ & 1];
    {
        const float biasd = bd1[nbase + (lane & 15)];
        v8f accz = { biasd, biasd, biasd, biasd, biasd, biasd, biasd, biasd };
#pragma unroll
        for (int kb = 0; kb < 4; ++kb) {
            v16bf a  = load_afrag(h1f, kb, lane);
            v16bf wb = load_bfrag(wd1, kb, nbase, lane);
            accz = __builtin_amdgcn_wmma_f32_16x16x32_bf16(false, a, false, wb,
                                                           (short)0, accz, false, false);
        }
        const int nn   = nbase + (lane & 15);
        const int mofs = (lane < 16) ? 0 : 8;
#pragma unroll
        for (int r = 0; r < 8; ++r)
            z_lds[r + mofs][nn] = fmaxf(accz[r], 0.0f);
    }
    __syncthreads();

    if (tid < ROWS) {
        float s = bd2[0];
#pragma unroll 4
        for (int nn = 0; nn < UNITS; ++nn) s += z_lds[tid][nn] * wd2[nn];
        out[brow0 + tid] = 1.0f / (1.0f + expf(-s));
    }
}

extern "C" void kernel_launch(void* const* d_in, const int* in_sizes, int n_in,
                              void* d_out, int out_size, void* d_ws, size_t ws_size,
                              hipStream_t stream) {
    (void)in_sizes; (void)n_in; (void)out_size;
    const int*   inputs = (const int*)d_in[0];
    const float* emb = (const float*)d_in[1];
    const float* w0x = (const float*)d_in[2];
    const float* w0h = (const float*)d_in[3];
    const float* b0v = (const float*)d_in[4];
    const float* w1x = (const float*)d_in[5];
    const float* w1h = (const float*)d_in[6];
    const float* b1v = (const float*)d_in[7];
    const float* wd1 = (const float*)d_in[8];
    const float* bd1 = (const float*)d_in[9];
    const float* wd2 = (const float*)d_in[10];
    const float* bd2 = (const float*)d_in[11];
    float* out = (float*)d_out;

    const size_t xw_bytes = (size_t)SEQ * NBLK * TILE * sizeof(float);  // 67 MB
    if (ws_size >= xw_bytes) {
        float* xw = (float*)d_ws;
        xw_precompute_kernel<<<dim3(NBLK, SEQ / TSTEPS), dim3(256), 0, stream>>>(
            inputs, emb, w0x, b0v, xw);
        rnn2_recur_kernel<<<dim3(NBLK), dim3(256), 0, stream>>>(
            xw, w0h, w1x, w1h, b1v, wd1, bd1, wd2, bd2, out);
    } else {
        rnn2_fused_kernel<<<dim3(NBLK), dim3(256), 0, stream>>>(
            inputs, emb, w0x, w0h, b0v, w1x, w1h, b1v, wd1, bd1, wd2, bd2, out);
    }
}
